// CrossRanker_43035572305965
// MI455X (gfx1250) — matrix operand order repaired and checked by reading, not compiled, not run on hardware
//
#include <hip/hip_runtime.h>

// ---------------------------------------------------------------------------
// CrossRanker for MI455X (gfx1250, wave32, WMMA + async global->LDS)
//   B=4, L=1024, S=8192, D=256, K_TOP=24
//   scores_avg[b,s] = mean_l softmax_s(Q.K^T/sqrt(D));  top-24 -> softmax ->
//   gather+scale keys.
// Strategy:
//   * one-shot fp32 -> bf16 hi/lo split into d_ws (~36.2 MB, fits 192MB L2)
//   * GEMM on v_wmma_f32_16x16x32_bf16, 3-term compensation (hi*hi+hi*lo+lo*hi)
//   * softmax without max-subtraction (logits are N(0,1)-scale: exp can't
//     overflow fp32) -> pass1 is a pure sum-exp, deterministic split-K combine
//   * shared operand staged per 8-wave block with GLOBAL_LOAD_ASYNC_TO_LDS_B128
//     (ASYNCcnt), double-buffered: tile t+1 streams into LDS while tile t's
//     24 WMMAs run; one s_wait_asynccnt + one barrier per tile.
// ---------------------------------------------------------------------------

#define B_     4
#define L_     1024
#define S_     8192
#define D_     256
#define KTOP_  24
#define SCALE_ 0.0625f       // 1/sqrt(256)
#define SSPLIT_ 8            // pass1 key-range splits (occupancy + determinism)

typedef __attribute__((ext_vector_type(16))) __bf16 v16bf;
typedef __attribute__((ext_vector_type(8)))  __bf16 v8bf;
typedef __attribute__((ext_vector_type(8)))  float  v8f;

#define SHUF16(a, b) __builtin_shufflevector(a, b, 0,1,2,3,4,5,6,7,8,9,10,11,12,13,14,15)

// ---- bf16 bit helpers (round-to-nearest-even) -----------------------------
static __device__ __forceinline__ unsigned short f2bf_bits(float f) {
    union { float f; unsigned u; } c; c.f = f;
    return (unsigned short)((c.u + 0x7FFFu + ((c.u >> 16) & 1u)) >> 16);
}
static __device__ __forceinline__ float bf_bits2f(unsigned short s) {
    union { unsigned u; float f; } o; o.u = ((unsigned)s) << 16;
    return o.f;
}

// ---- CDNA5 async global->LDS (16B per lane, per-lane LDS scatter) ---------
// LDS destination is the 32-bit LDS offset.
static __device__ __forceinline__ unsigned lds_off_of(const void* p) {
    return (unsigned)(unsigned long long)
        (__attribute__((address_space(3))) const char*)p;
}
static __device__ __forceinline__ void async_ld16(unsigned lds, const unsigned short* g) {
    asm volatile("global_load_async_to_lds_b128 %0, %1, off"
                 :: "v"(lds), "v"(g)
                 : "memory");
}
static __device__ __forceinline__ void wait_async0() {
    asm volatile("s_wait_asynccnt 0x0" ::: "memory");
}

// ---- 3-term compensated bf16 WMMA: (ah+al)*(bh+bl) ~= ah*bh + ah*bl + al*bh
static __device__ __forceinline__ v8f mma3(v16bf ah, v16bf al, v16bf bh, v16bf bl, v8f c) {
    c = __builtin_amdgcn_wmma_f32_16x16x32_bf16(false, ah, false, bh, (short)0, c, false, false);
    c = __builtin_amdgcn_wmma_f32_16x16x32_bf16(false, ah, false, bl, (short)0, c, false, false);
    c = __builtin_amdgcn_wmma_f32_16x16x32_bf16(false, al, false, bh, (short)0, c, false, false);
    return c;
}

// ===========================================================================
// Kernel 0: fp32 -> (bf16 hi, bf16 lo) split, 4 elements/thread.
// ===========================================================================
__global__ __launch_bounds__(256) void cr_convert(const float* __restrict__ src,
                                                  unsigned short* __restrict__ hi,
                                                  unsigned short* __restrict__ lo,
                                                  int n4) {
    const int i = blockIdx.x * 256 + threadIdx.x;
    if (i >= n4) return;
    const float4 x = reinterpret_cast<const float4*>(src)[i];
    const float f[4] = {x.x, x.y, x.z, x.w};
    unsigned short hh[4], ll[4];
#pragma unroll
    for (int e = 0; e < 4; ++e) {
        hh[e] = f2bf_bits(f[e]);
        ll[e] = f2bf_bits(f[e] - bf_bits2f(hh[e]));
    }
    reinterpret_cast<ushort4*>(hi)[i] = make_ushort4(hh[0], hh[1], hh[2], hh[3]);
    reinterpret_cast<ushort4*>(lo)[i] = make_ushort4(ll[0], ll[1], ll[2], ll[3]);
}

// ===========================================================================
// Pass 1: Zpart[chunk][b*L+row] = sum over this chunk's keys of exp(z).
// Block = 8 waves, each wave owns one 16-row query tile (A hi/lo register-
// resident, 128 VGPRs); the 16-key B tile is async-staged into a double-
// buffered LDS fragment store and shared by all 8 waves.
// grid = (B * L/128 row-groups) * SSPLIT = 256 blocks.
// ===========================================================================
__global__ __launch_bounds__(256) void cr_pass1(
        const unsigned short* __restrict__ Qhi, const unsigned short* __restrict__ Qlo,
        const unsigned short* __restrict__ Khi, const unsigned short* __restrict__ Klo,
        float* __restrict__ Zpart) {
    const int rg    = blockIdx.x >> 3;         // row group (0..31)
    const int chunk = blockIdx.x & 7;          // key-range split
    const int b     = rg >> 3;
    const int mg0   = (rg & 7) * 128;
    const int wave  = threadIdx.x >> 5;
    const int lane  = threadIdx.x & 31;
    const int half  = lane >> 4;
    const int l16   = lane & 15;
    const int m0    = mg0 + wave * 16;

    // Register-resident A fragments (contiguous K order: lane half h holds
    // K {16h..16h+15}, matching the staged B fragments).
    const size_t qoff = ((size_t)b * L_ + (size_t)(m0 + l16)) * D_;
    v16bf aHi[8], aLo[8];
#pragma unroll
    for (int kb = 0; kb < 8; ++kb) {
        const size_t o = qoff + (size_t)(kb * 32 + half * 16);
        v8bf h0 = *(const v8bf*)(Qhi + o);
        v8bf h1 = *(const v8bf*)(Qhi + o + 8);
        v8bf l0 = *(const v8bf*)(Qlo + o);
        v8bf l1 = *(const v8bf*)(Qlo + o + 8);
        aHi[kb] = SHUF16(h0, h1);
        aLo[kb] = SHUF16(l0, l1);
    }

    float zsum[8];
#pragma unroll
    for (int r = 0; r < 8; ++r) zsum[r] = 0.0f;

    // Double-buffered B fragment store; thread t stages (kb=t>>5, lanepos=t&31).
    __shared__ v16bf sBhi[2][256];
    __shared__ v16bf sBlo[2][256];
    unsigned ldsHi[2], ldsLo[2];
    ldsHi[0] = lds_off_of(&sBhi[0][threadIdx.x]);
    ldsHi[1] = lds_off_of(&sBhi[1][threadIdx.x]);
    ldsLo[0] = lds_off_of(&sBlo[0][threadIdx.x]);
    ldsLo[1] = lds_off_of(&sBlo[1][threadIdx.x]);

    const int skb   = threadIdx.x >> 5;
    const int sl16  = threadIdx.x & 15;
    const int shalf = (threadIdx.x >> 4) & 1;
    const int tiles = (S_ / 16) / SSPLIT_;      // 64 key tiles per block
    const int tb    = chunk * tiles;

    // prologue: stage tile 0 into buffer 0 (16 consecutive K per lane-pos)
    {
        const size_t o = ((size_t)b * S_ + (size_t)(((tb) << 4) + sl16)) * D_
                       + (size_t)(skb * 32 + shalf * 16);
        async_ld16(ldsHi[0], Khi + o); async_ld16(ldsHi[0] + 16, Khi + o + 8);
        async_ld16(ldsLo[0], Klo + o); async_ld16(ldsLo[0] + 16, Klo + o + 8);
    }
    for (int t = 0; t < tiles; ++t) {
        const int cur = t & 1;
        wait_async0();                          // own staging portion landed
        __syncthreads();                        // all portions landed; prev reads done
        if (t + 1 < tiles) {                    // prefetch next tile during compute
            const int nxt = cur ^ 1;
            const size_t o = ((size_t)b * S_ + (size_t)(((tb + t + 1) << 4) + sl16)) * D_
                           + (size_t)(skb * 32 + shalf * 16);
            async_ld16(ldsHi[nxt], Khi + o); async_ld16(ldsHi[nxt] + 16, Khi + o + 8);
            async_ld16(ldsLo[nxt], Klo + o); async_ld16(ldsLo[nxt] + 16, Klo + o + 8);
        }
        v8f acc = {};
#pragma unroll
        for (int kb = 0; kb < 8; ++kb)
            acc = mma3(aHi[kb], aLo[kb], sBhi[cur][kb * 32 + lane],
                       sBlo[cur][kb * 32 + lane], acc);
#pragma unroll
        for (int r = 0; r < 8; ++r)
            zsum[r] += __expf(acc[r] * SCALE_);
    }

    // Fold the 16 columns of each half (C layout: VGPR r -> row half*8+r).
#pragma unroll
    for (int r = 0; r < 8; ++r) {
#pragma unroll
        for (int msk = 1; msk < 16; msk <<= 1)
            zsum[r] += __shfl_xor(zsum[r], msk, 32);
    }
    if (l16 == 0) {
#pragma unroll
        for (int r = 0; r < 8; ++r)
            Zpart[(size_t)chunk * (B_ * L_) + (size_t)b * L_ + m0 + half * 8 + r] = zsum[r];
    }
}

// ===========================================================================
// Combine pass-1 partials (fixed order -> deterministic), store 1/Z.
// ===========================================================================
__global__ __launch_bounds__(256) void cr_zcombine(const float* __restrict__ Zpart,
                                                   float* __restrict__ zInv) {
    const int i = blockIdx.x * 256 + threadIdx.x;
    if (i >= B_ * L_) return;
    float s = 0.0f;
#pragma unroll
    for (int c = 0; c < SSPLIT_; ++c) s += Zpart[(size_t)c * (B_ * L_) + i];
    zInv[i] = 1.0f / s;
}

// ===========================================================================
// Pass 2: scores_avg[b,s] = (1/L) * sum_l exp(z[l,s]) * zInv[l].
// Block = 8 waves, each wave owns one 16-key tile (B hi/lo register-resident);
// the 16-query A tile is async-staged (double-buffered) and shared.
// grid = B * S/128 = 256 blocks.
// ===========================================================================
__global__ __launch_bounds__(256) void cr_pass2(
        const unsigned short* __restrict__ Qhi, const unsigned short* __restrict__ Qlo,
        const unsigned short* __restrict__ Khi, const unsigned short* __restrict__ Klo,
        const float* __restrict__ zInv, float* __restrict__ avg) {
    const int b     = blockIdx.x >> 6;
    const int chunk = blockIdx.x & 63;
    const int wave  = threadIdx.x >> 5;
    const int lane  = threadIdx.x & 31;
    const int half  = lane >> 4;
    const int l16   = lane & 15;
    const int n0    = ((chunk << 3) + wave) << 4;

    // Register-resident B fragments for this wave's 16 keys.
    const size_t koff0 = ((size_t)b * S_ + (size_t)(n0 + l16)) * D_;
    v16bf bHi[8], bLo[8];
#pragma unroll
    for (int kb = 0; kb < 8; ++kb) {
        const size_t o = koff0 + (size_t)(kb * 32 + half * 16);
        bHi[kb] = *(const v16bf*)(Khi + o);
        bLo[kb] = *(const v16bf*)(Klo + o);
    }

    __shared__ v16bf sAhi[2][256];
    __shared__ v16bf sAlo[2][256];
    unsigned ldsHi[2], ldsLo[2];
    ldsHi[0] = lds_off_of(&sAhi[0][threadIdx.x]);
    ldsHi[1] = lds_off_of(&sAhi[1][threadIdx.x]);
    ldsLo[0] = lds_off_of(&sAlo[0][threadIdx.x]);
    ldsLo[1] = lds_off_of(&sAlo[1][threadIdx.x]);

    const int skb   = threadIdx.x >> 5;
    const int sl16  = threadIdx.x & 15;
    const int shalf = (threadIdx.x >> 4) & 1;

    // prologue: stage query tile m0=0 (A fragment = 16 consecutive K per lane-pos)
    {
        const size_t o = ((size_t)b * L_ + (size_t)sl16) * D_
                       + (size_t)(skb * 32 + shalf * 16);
        async_ld16(ldsHi[0], Qhi + o); async_ld16(ldsHi[0] + 16, Qhi + o + 8);
        async_ld16(ldsLo[0], Qlo + o); async_ld16(ldsLo[0] + 16, Qlo + o + 8);
    }
    float colacc = 0.0f;
    for (int t = 0; t < L_ / 16; ++t) {
        const int m0  = t << 4;
        const int cur = t & 1;
        wait_async0();
        __syncthreads();
        if (t + 1 < L_ / 16) {
            const int nxt = cur ^ 1;
            const size_t o = ((size_t)b * L_ + (size_t)(m0 + 16 + sl16)) * D_
                           + (size_t)(skb * 32 + shalf * 16);
            async_ld16(ldsHi[nxt], Qhi + o); async_ld16(ldsHi[nxt] + 16, Qhi + o + 8);
            async_ld16(ldsLo[nxt], Qlo + o); async_ld16(ldsLo[nxt] + 16, Qlo + o + 8);
        }
        v8f acc = {};
#pragma unroll
        for (int kb = 0; kb < 8; ++kb)
            acc = mma3(sAhi[cur][kb * 32 + lane], sAlo[cur][kb * 32 + lane],
                       bHi[kb], bLo[kb], acc);

        const size_t rb = (size_t)b * L_ + m0 + half * 8;
#pragma unroll
        for (int r = 0; r < 8; ++r)
            colacc += __expf(acc[r] * SCALE_) * zInv[rb + r];
    }
    colacc += __shfl_xor(colacc, 16, 32);      // fold the two row halves
    if (half == 0)
        avg[(size_t)b * S_ + n0 + l16] = colacc * (1.0f / (float)L_);
}

// ===========================================================================
// Pass 3: per batch row, top-24 of scores_avg (ties -> lowest index, matching
// jax.lax.top_k), softmax the 24, gather+scale fp32 keys.
// ===========================================================================
__global__ __launch_bounds__(256) void cr_pass3(const float* __restrict__ Kd,
                                                const float* __restrict__ avg,
                                                float* __restrict__ out) {
    const int b   = blockIdx.x;
    const int tid = threadIdx.x;
    __shared__ float vals[S_];                  // 32 KB of 320 KB LDS
    __shared__ float rv[256];
    __shared__ int   ri[256];
    __shared__ float topv[KTOP_];
    __shared__ int   topi[KTOP_];
    __shared__ float tw[KTOP_];

    for (int i = tid; i < S_; i += 256) vals[i] = avg[(size_t)b * S_ + i];

    for (int k = 0; k < KTOP_; ++k) {
        __syncthreads();
        float bv = -3.0e38f; int bi = 0x7fffffff;
        for (int i = tid; i < S_; i += 256) {
            float v = vals[i];
            if (v > bv || (v == bv && i < bi)) { bv = v; bi = i; }
        }
        rv[tid] = bv; ri[tid] = bi;
        __syncthreads();
        for (int s = 128; s > 0; s >>= 1) {
            if (tid < s) {
                if (rv[tid + s] > rv[tid] ||
                    (rv[tid + s] == rv[tid] && ri[tid + s] < ri[tid])) {
                    rv[tid] = rv[tid + s]; ri[tid] = ri[tid + s];
                }
            }
            __syncthreads();
        }
        if (tid == 0) {
            topv[k] = rv[0];
            topi[k] = ri[0];
            vals[ri[0]] = -3.0e38f;
        }
    }
    __syncthreads();
    if (tid == 0) {
        float m = topv[0];
        float s = 0.0f;
        for (int k = 0; k < KTOP_; ++k) { float e = __expf(topv[k] - m); tw[k] = e; s += e; }
        float inv = 1.0f / s;
        for (int k = 0; k < KTOP_; ++k) tw[k] *= inv;
    }
    __syncthreads();
    for (int i = tid; i < KTOP_ * D_; i += 256) {
        const int k = i / D_, d = i - k * D_;
        out[(size_t)b * KTOP_ * D_ + i] =
            Kd[((size_t)b * S_ + (size_t)topi[k]) * D_ + d] * tw[k];
    }
}

// ===========================================================================
// Workspace layout (bytes):
//   Qhi 2MB | Qlo 2MB | Khi 16MB | Klo 16MB | Zpart 128KB | zInv 16KB  ~36.2MB
// ===========================================================================
extern "C" void kernel_launch(void* const* d_in, const int* in_sizes, int n_in,
                              void* d_out, int out_size, void* d_ws, size_t ws_size,
                              hipStream_t stream) {
    (void)in_sizes; (void)n_in; (void)out_size; (void)ws_size;
    const float* Q  = (const float*)d_in[0];    // [B, L, D] fp32
    const float* Kd = (const float*)d_in[1];    // [B, S, D] fp32
    float* out = (float*)d_out;                 // [B, KTOP, D] then [B, S]
    float* avg = out + (size_t)B_ * KTOP_ * D_;

    unsigned short* Qhi = (unsigned short*)d_ws;
    unsigned short* Qlo = Qhi + (size_t)B_ * L_ * D_;
    unsigned short* Khi = Qlo + (size_t)B_ * L_ * D_;
    unsigned short* Klo = Khi + (size_t)B_ * S_ * D_;
    float* Zpart = (float*)(Klo + (size_t)B_ * S_ * D_);
    float* zInv  = Zpart + (size_t)SSPLIT_ * B_ * L_;

    const int nQ4 = B_ * L_ * D_ / 4;           // 262144 vec4 elems
    const int nK4 = B_ * S_ * D_ / 4;           // 2097152 vec4 elems
    cr_convert<<<dim3(nQ4 / 256), 256, 0, stream>>>(Q,  Qhi, Qlo, nQ4);
    cr_convert<<<dim3(nK4 / 256), 256, 0, stream>>>(Kd, Khi, Klo, nK4);
    cr_pass1<<<dim3(B_ * (L_ / 128) * SSPLIT_), 256, 0, stream>>>(Qhi, Qlo, Khi, Klo, Zpart);
    cr_zcombine<<<dim3((B_ * L_ + 255) / 256), 256, 0, stream>>>(Zpart, zInv);
    cr_pass2<<<dim3(B_ * (S_ / 128)), 256, 0, stream>>>(Qhi, Qlo, Khi, Klo, zInv, avg);
    cr_pass3<<<dim3(B_), 256, 0, stream>>>(Kd, avg, out);
}
